// VanillaRNN_62148176773447
// MI455X (gfx1250) — compile-verified, matrix operand order checked by reading, not run on hardware
//
#include <hip/hip_runtime.h>

typedef __attribute__((ext_vector_type(16))) _Float16 v16h;
typedef __attribute__((ext_vector_type(8)))  _Float16 v8h;
typedef __attribute__((ext_vector_type(8)))  float    v8f;

#define HDIM 512
#define SDIM 512
#define CDIM 10
#define BT   16      // batch rows per workgroup (one WMMA M-tile)
#define HS   520     // padded LDS stride (halves) for h tile
#define BSTR 520     // padded LDS stride (halves) for LDS-resident B rows

// W_hh fp32 [k][j] -> Wt fp16 transposed [j][k] (WMMA-B friendly: per-lane
// fragment = 16 consecutive k for a fixed output column j, contiguous 32B)
__global__ __launch_bounds__(256) void prep_wt(const float* __restrict__ whh,
                                               _Float16* __restrict__ wt) {
  int idx = blockIdx.x * 256 + threadIdx.x;   // 0 .. 512*512-1
  int j = idx >> 9;
  int k = idx & (HDIM - 1);
  wt[(size_t)j * HDIM + k] = (_Float16)whh[(size_t)k * HDIM + j];
}

static __device__ __forceinline__ v16h ld_frag16(const _Float16* p) {
  const v8h* q = (const v8h*)p;                 // 16B-aligned
  return __builtin_shufflevector(q[0], q[1],
          0,1,2,3,4,5,6,7,8,9,10,11,12,13,14,15);
}

__global__ __launch_bounds__(256, 1) void rnn_kernel(
    const float* __restrict__ x,    const float* __restrict__ whx_g,
    const _Float16* __restrict__ wt, const float* __restrict__ wph,
    const float* __restrict__ bh_g, const float* __restrict__ bp,
    float* __restrict__ out)
{
  extern __shared__ char smem[];
  _Float16* hbuf = (_Float16*)smem;                       // BT*HS halves
  _Float16* blds = hbuf + BT * HS;                        // 256*BSTR halves (16 B-tiles)
  _Float16* xT   = blds + 256 * BSTR;                     // SDIM*BT halves (x, fp16)
  float*    whx  = (float*)(xT + SDIM * BT);              // HDIM
  float*    bh   = whx + HDIM;                            // HDIM

  const int tid = threadIdx.x;
  const int b0  = blockIdx.x * BT;

  // h := 0
  for (int i = tid; i < BT * HS; i += 256) hbuf[i] = (_Float16)0.f;
  // stage x transposed (fp16): xT[t][row]
  for (int i = tid; i < BT * SDIM; i += 256) {
    int row = i >> 9;
    int t   = i & (SDIM - 1);
    xT[t * BT + row] = (_Float16)x[(size_t)(b0 + row) * SDIM + t];
  }
  for (int i = tid; i < HDIM; i += 256) { whx[i] = whx_g[i]; bh[i] = bh_g[i]; }

  // stage LDS-resident half of Wt: for wave w, tiles j=2,3 -> lds tiles w*2+{0,1}
  // lds tile lt (0..15) row rr holds Wt row ((lt>>1)*4 + 2 + (lt&1))*16 + rr
  for (int c = tid; c < 256 * 64; c += 256) {   // 16B chunks: 256 rows * 64 chunks
    int r   = c >> 6;            // 0..255 packed lds row
    int off = (c & 63) * 8;      // halve offset within row
    int lt  = r >> 4;
    int rr  = r & 15;
    int grow = ((lt >> 1) * 4 + 2 + (lt & 1)) * 16 + rr;
    *(v8h*)(blds + (size_t)r * BSTR + off) =
        *(const v8h*)(wt + (size_t)grow * HDIM + off);
  }

  const int lane = tid & 31;
  const int w    = tid >> 5;      // wave 0..7
  const int lo   = lane & 15;     // WMMA lane column / A row
  const int hi   = lane >> 4;     // WMMA K-half selector

  // ---- Register-resident B: N-tiles j=0,1 (2 * 16 * v16h = 256 VGPRs). ----
  v16h Breg[2][16];
  #pragma unroll
  for (int j = 0; j < 2; ++j) {
    const _Float16* bptr = wt + (size_t)((w * 4 + j) * 16 + lo) * HDIM + hi * 16;
    #pragma unroll
    for (int k = 0; k < 16; ++k)
      Breg[j][k] = *(const v16h*)(bptr + k * 32);
  }

  __syncthreads();

  // Per-wave LDS base pointers
  const _Float16* arow  = hbuf + lo * HS + hi * 16;
  const _Float16* b2row = blds + (size_t)((w * 2 + 0) * 16 + lo) * BSTR + hi * 16;
  const _Float16* b3row = blds + (size_t)((w * 2 + 1) * 16 + lo) * BSTR + hi * 16;

  for (int t = 0; t < SDIM; ++t) {
    v8f acc[4] = {};
    #pragma unroll
    for (int k = 0; k < 16; ++k) {
      v16h A  = ld_frag16(arow  + k * 32);
      v16h B2 = ld_frag16(b2row + k * 32);
      v16h B3 = ld_frag16(b3row + k * 32);
      acc[0] = __builtin_amdgcn_wmma_f32_16x16x32_f16(
                   false, A, false, Breg[0][k], (short)0, acc[0], false, false);
      acc[1] = __builtin_amdgcn_wmma_f32_16x16x32_f16(
                   false, A, false, Breg[1][k], (short)0, acc[1], false, false);
      acc[2] = __builtin_amdgcn_wmma_f32_16x16x32_f16(
                   false, A, false, B2, (short)0, acc[2], false, false);
      acc[3] = __builtin_amdgcn_wmma_f32_16x16x32_f16(
                   false, A, false, B3, (short)0, acc[3], false, false);
    }
    __syncthreads();  // all waves done reading h -> safe to overwrite

    #pragma unroll
    for (int j = 0; j < 4; ++j) {
      const int   n  = (w * 4 + j) * 16 + lo;
      const float wx = whx[n];
      const float bb = bh[n];
      #pragma unroll
      for (int r = 0; r < 8; ++r) {
        const int m = hi * 8 + r;                     // local batch row
        float y = acc[j][r] + (float)xT[t * BT + m] * wx + bb;
        y = fminf(fmaxf(y, -15.f), 15.f);
        float e = __expf(2.f * y);                    // tanh via exp
        hbuf[m * HS + n] = (_Float16)((e - 1.f) / (e + 1.f));
      }
    }
    __syncthreads();  // new h visible before next step's A loads
  }

  // Output projection: out[b0+row, c] = h[row,:] @ W_ph[:,c] + b_p[c]
  if (tid < BT * CDIM) {
    int row = tid / CDIM, c = tid % CDIM;
    float acc = bp[c];
    for (int k = 0; k < HDIM; ++k)
      acc += (float)hbuf[row * HS + k] * wph[(size_t)k * CDIM + c];
    out[(size_t)(b0 + row) * CDIM + c] = acc;
  }
}

extern "C" void kernel_launch(void* const* d_in, const int* in_sizes, int n_in,
                              void* d_out, int out_size, void* d_ws, size_t ws_size,
                              hipStream_t stream) {
  const float* x   = (const float*)d_in[0];
  const float* whx = (const float*)d_in[1];
  const float* whh = (const float*)d_in[2];
  const float* wph = (const float*)d_in[3];
  const float* bh  = (const float*)d_in[4];
  const float* bpv = (const float*)d_in[5];

  _Float16* wt = (_Float16*)d_ws;   // 512*512*2 = 512 KB scratch

  prep_wt<<<(HDIM * HDIM) / 256, 256, 0, stream>>>(whh, wt);

  size_t smem = (size_t)BT * HS * 2        // h tile (fp16)
              + (size_t)256 * BSTR * 2     // LDS-resident B tiles
              + (size_t)SDIM * BT * 2      // x transposed (fp16)
              + (size_t)HDIM * 4 * 2;      // whx + bh (fp32)
  rnn_kernel<<<1024 / BT, 256, smem, stream>>>(x, whx, wt, wph, bh, bpv,
                                               (float*)d_out);
}
